// SlidingWindowAttention_2164663517569
// MI455X (gfx1250) — matrix operand order repaired and checked by reading, not compiled, not run on hardware
//
#include <hip/hip_runtime.h>
#include <hip/hip_bf16.h>

// ---------------------------------------------------------------------------
// Sliding-window GQA attention block for MI455X (gfx1250), wave32 + WMMA bf16.
//   1) fp32 -> bf16 conversion of x and the 4 weight matrices (workspace)
//   2) Q = Xb @ Wq^T + bq   [4096 x 1024]  (WMMA GEMM, async-LDS B staging)
//   3) K = Xb @ Wk^T + bk   [4096 x  256]
//   4) V = Xb @ Wv^T + bv   -> Vt[256, 4096] (transposed store)
//   5) flash-style sliding-window attention with sink softmax -> AO bf16
//   6) out = AO @ Wo^T + bo (fp32)
// GEMM: 4-wave workgroups, 128x64 macro-tile; B stripe staged into LDS with
// global_load_async_to_lds_b128 (double-buffered, ASYNCcnt), 4x weight reuse.
// ---------------------------------------------------------------------------

typedef __bf16 bf16_t;
typedef __attribute__((ext_vector_type(16))) __bf16 v16bf;
typedef __attribute__((ext_vector_type(8)))  __bf16 v8bf;
typedef __attribute__((ext_vector_type(8)))  float  v8f;

#define S_LEN    4096
#define D_MODEL  1024
#define N_HEADS  8
#define N_KV     2
#define HEAD_DIM 128
#define WIN_SZ   256

#define BM 128          // GEMM macro-tile M (4 waves x 32 rows)
#define BN 64           // GEMM macro-tile N
#define BSTRIDE 40      // padded LDS row stride (elements) for the B stripe

// --- A-fragment (16x32 bf16, row-major source).
// lane m = lane&15 ; elems 0..15 -> K = kb..kb+15
// with kb = 16*(lane>>4).  Two contiguous 16-byte loads per lane.
__device__ __forceinline__ v16bf load_a_frag(const bf16_t* p, int ld, int row0,
                                             int k0, int lane) {
  int r  = row0 + (lane & 15);
  int kb = k0 + ((lane >> 4) << 4);
  if (r  < 0) r  = 0;   // clamped lanes are always multiplied by 0 / masked
  if (kb < 0) kb = 0;
  const bf16_t* base = p + (size_t)r * ld + kb;
  union { v16bf v; v8bf h[2]; } u;
  u.h[0] = *(const v8bf*)(base);
  u.h[1] = *(const v8bf*)(base + 8);
  return u.v;
}

// --- B-fragment (32x16 bf16): lane n = lane&15 holds K = 16*(lane>>4)+0..15
// (one contiguous 32-byte run). `p` rows are B columns (K^T / Vt / W qualify).
__device__ __forceinline__ v16bf load_b_frag(const bf16_t* p, int ld, int col0,
                                             int k0, int lane) {
  int c  = col0 + (lane & 15);
  int kb = k0 + ((lane >> 4) << 4);
  if (c  < 0) c  = 0;
  if (kb < 0) kb = 0;
  const bf16_t* base = p + (size_t)c * ld + kb;
  union { v16bf v; v8bf h[2]; } u;
  u.h[0] = *(const v8bf*)(base);
  u.h[1] = *(const v8bf*)(base + 8);
  return u.v;
}

__device__ __forceinline__ v8f wmma_bf16(v16bf a, v16bf b, v8f c) {
  return __builtin_amdgcn_wmma_f32_16x16x32_bf16(false, a, false, b,
                                                 (short)0, c, false, false);
}

// ---------------------------------------------------------------------------
__global__ void __launch_bounds__(256)
f32_to_bf16_kernel(const float* __restrict__ src, bf16_t* __restrict__ dst, int n) {
  int i = blockIdx.x * blockDim.x + threadIdx.x;
  if (i < n) dst[i] = (bf16_t)src[i];
}

// ---------------------------------------------------------------------------
// C[M,N] = A[M,K] @ W[N,K]^T + bias.
// 4 waves / WG; wave w owns rows [m0+32w, m0+32w+32) x all 64 N columns.
// B stripe (64 cols x 32 k, 4 KB) staged to LDS by async loads, double-buffered.
// mode 0: bf16 row-major out | mode 1: bf16 transposed out | mode 2: fp32 out
__global__ void __launch_bounds__(128)
gemm_bf16_wmma(const bf16_t* __restrict__ A, const bf16_t* __restrict__ W,
               const float* __restrict__ bias, void* __restrict__ out,
               int M, int N, int K, int mode, int ldo) {
  const int tid     = threadIdx.x;
  const int lane    = tid & 31;
  const int wave    = tid >> 5;
  const int nTilesN = N / BN;
  const int m0      = (int)(blockIdx.x / nTilesN) * BM;
  const int n0      = (int)(blockIdx.x % nTilesN) * BN;
  const int mw      = m0 + wave * 32;

  __shared__ __align__(16) bf16_t Bsh[2][BN * BSTRIDE];

  const v8f vzero = {0.f, 0.f, 0.f, 0.f, 0.f, 0.f, 0.f, 0.f};
  v8f acc[2][4];
#pragma unroll
  for (int s = 0; s < 2; ++s)
#pragma unroll
    for (int t = 0; t < 4; ++t) acc[s][t] = vzero;

  // async-stage one 64x32 B stripe (256 x 16B chunks; 2 per thread)
  auto stageB = [&](int k0, int buf) {
#pragma unroll
    for (int u = 0; u < 2; ++u) {
      const int c   = tid * 2 + u;          // chunk 0..255
      const int row = c >> 2;               // B column / W row
      const int seg = c & 3;                // 16B segment in the 64B k-run
      const bf16_t* g = W + (size_t)(n0 + row) * K + k0 + seg * 8;
      const unsigned ldsoff =
          (unsigned)(uintptr_t)(&Bsh[buf][row * BSTRIDE + seg * 8]);
      asm volatile("global_load_async_to_lds_b128 %0, %1, off"
                   :: "v"(ldsoff), "v"(g) : "memory");
    }
  };

  stageB(0, 0);
  int cur = 0;
  for (int k0 = 0; k0 < K; k0 += 32) {
    asm volatile("s_wait_asynccnt 0x0" ::: "memory");
    __syncthreads();                        // B[cur] ready; B[cur^1] drained
    if (k0 + 32 < K) stageB(k0 + 32, cur ^ 1);

    v16bf a0 = load_a_frag(A, K, mw,      k0, lane);
    v16bf a1 = load_a_frag(A, K, mw + 16, k0, lane);
    if (k0 + 32 < K)  // gfx1250 global_prefetch_b8 of the next A stripe
      __builtin_prefetch(A + (size_t)(mw + (lane & 15)) * K + k0 + 32, 0, 1);

#pragma unroll
    for (int t = 0; t < 4; ++t) {
      const bf16_t* bp =
          &Bsh[cur][(t * 16 + (lane & 15)) * BSTRIDE + ((lane >> 4) << 4)];
      union { v16bf v; v8bf h[2]; } ub;
      ub.h[0] = *(const v8bf*)(bp);
      ub.h[1] = *(const v8bf*)(bp + 8);
      acc[0][t] = wmma_bf16(a0, ub.v, acc[0][t]);
      acc[1][t] = wmma_bf16(a1, ub.v, acc[1][t]);
    }
    cur ^= 1;
  }

  const int col   = lane & 15;
  const int rbase = (lane >> 4) << 3;       // lanes 16..31 own rows 8..15
#pragma unroll
  for (int s = 0; s < 2; ++s) {
#pragma unroll
    for (int t = 0; t < 4; ++t) {
      const int   n  = n0 + t * 16 + col;
      const float bv = bias ? bias[n] : 0.0f;
#pragma unroll
      for (int r = 0; r < 8; ++r) {
        const int   m = mw + s * 16 + rbase + r;
        const float v = acc[s][t][r] + bv;
        if (mode == 0)      ((bf16_t*)out)[(size_t)m * ldo + n] = (bf16_t)v;
        else if (mode == 1) ((bf16_t*)out)[(size_t)n * ldo + m] = (bf16_t)v;
        else                ((float*)out)[(size_t)m * ldo + n]  = v;
      }
    }
  }
}

// ---------------------------------------------------------------------------
// Flash-style sliding-window attention, one wave per (head, 16-query block).
// Q : [S, 1024] bf16 (col = h*128+d)   K : [S, 256] bf16   Vt : [256, S] bf16
__global__ void __launch_bounds__(32)
swa_attn_wmma(const bf16_t* __restrict__ Q, const bf16_t* __restrict__ Kmat,
              const bf16_t* __restrict__ Vt, const float* __restrict__ sinkp,
              bf16_t* __restrict__ AO) {
  const int lane = threadIdx.x & 31;
  const int nQB  = S_LEN / 16;
  const int h    = (int)blockIdx.x / nQB;
  const int q0   = ((int)blockIdx.x % nQB) * 16;
  const int hk   = h / (N_HEADS / N_KV);
  const float sink  = sinkp[0];
  const float scale = 0.08838834764831845f;  // 1/sqrt(128)

  __shared__ __align__(16) bf16_t Pl[16 * 32];  // P-tile C->A relayout buffer

  const v8f vzero = {0.f, 0.f, 0.f, 0.f, 0.f, 0.f, 0.f, 0.f};

  v16bf qf[4];                 // Q fragments, d = 0..127 (reused all 9 steps)
#pragma unroll
  for (int kc = 0; kc < 4; ++kc)
    qf[kc] = load_a_frag(Q, N_HEADS * HEAD_DIM, q0, h * HEAD_DIM + kc * 32, lane);

  v8f o[8];                    // 16x128 fp32 output accumulator
#pragma unroll
  for (int d = 0; d < 8; ++d) o[d] = vzero;
  float mrow[8], lrow[8];
#pragma unroll
  for (int r = 0; r < 8; ++r) { mrow[r] = sink; lrow[r] = 0.0f; }

  const int col   = lane & 15;
  const int rbase = (lane >> 4) << 3;

  // key span [q0-255, q0+15] padded to 9 aligned steps of 32 (extras masked)
  for (int kt = q0 - 272; kt < q0; kt += 32) {
    v8f sc0 = vzero, sc1 = vzero;
#pragma unroll
    for (int kc = 0; kc < 4; ++kc) {
      v16bf b0 = load_b_frag(Kmat + hk * HEAD_DIM, N_KV * HEAD_DIM, kt,      kc * 32, lane);
      v16bf b1 = load_b_frag(Kmat + hk * HEAD_DIM, N_KV * HEAD_DIM, kt + 16, kc * 32, lane);
      sc0 = wmma_bf16(qf[kc], b0, sc0);
      sc1 = wmma_bf16(qf[kc], b1, sc1);
    }

    float p0[8], p1[8];
#pragma unroll
    for (int r = 0; r < 8; ++r) {
      const int i  = q0 + rbase + r;
      const int j0 = kt + col, j1 = kt + 16 + col;
      float v0 = sc0[r] * scale, v1 = sc1[r] * scale;
      v0 = (j0 >= 0 && j0 <= i && j0 > i - WIN_SZ) ? v0 : -65504.0f;
      v1 = (j1 >= 0 && j1 <= i && j1 > i - WIN_SZ) ? v1 : -65504.0f;
      float rmax = fmaxf(v0, v1);
#pragma unroll
      for (int off = 1; off < 16; off <<= 1)   // stays within 16-lane half
        rmax = fmaxf(rmax, __shfl_xor(rmax, off, 32));
      const float mnew = fmaxf(mrow[r], rmax);
      const float f    = __expf(mrow[r] - mnew);
      const float e0   = __expf(v0 - mnew);
      const float e1   = __expf(v1 - mnew);
      float rs = e0 + e1;
#pragma unroll
      for (int off = 1; off < 16; off <<= 1)
        rs += __shfl_xor(rs, off, 32);
      lrow[r] = lrow[r] * f + rs;
      mrow[r] = mnew;
      p0[r] = e0; p1[r] = e1;
#pragma unroll
      for (int d = 0; d < 8; ++d) o[d][r] *= f;
    }

    __syncthreads();
#pragma unroll
    for (int r = 0; r < 8; ++r) {
      const int row = rbase + r;
      Pl[row * 32 + col]      = (bf16_t)p0[r];
      Pl[row * 32 + 16 + col] = (bf16_t)p1[r];
    }
    __syncthreads();
    union { v16bf v; v8bf hh[2]; } pu;
    {
      const bf16_t* bp = Pl + (lane & 15) * 32 + ((lane >> 4) << 4);
      pu.hh[0] = *(const v8bf*)(bp);
      pu.hh[1] = *(const v8bf*)(bp + 8);
    }

#pragma unroll
    for (int d = 0; d < 8; ++d) {
      v16bf bv = load_b_frag(Vt, S_LEN, hk * HEAD_DIM + d * 16, kt, lane);
      o[d] = wmma_bf16(pu.v, bv, o[d]);
    }
  }

#pragma unroll
  for (int r = 0; r < 8; ++r) {
    const float inv = 1.0f / (lrow[r] + __expf(sink - mrow[r]));
    const int   row = q0 + rbase + r;
#pragma unroll
    for (int d = 0; d < 8; ++d)
      AO[(size_t)row * (N_HEADS * HEAD_DIM) + h * HEAD_DIM + d * 16 + col] =
          (bf16_t)(o[d][r] * inv);
  }
}

// ---------------------------------------------------------------------------
extern "C" void kernel_launch(void* const* d_in, const int* in_sizes, int n_in,
                              void* d_out, int out_size, void* d_ws, size_t ws_size,
                              hipStream_t stream) {
  (void)in_sizes; (void)n_in; (void)out_size; (void)ws_size;
  const float* x    = (const float*)d_in[0];
  const float* sink = (const float*)d_in[1];
  const float* Wq   = (const float*)d_in[2];
  const float* bq   = (const float*)d_in[3];
  const float* Wk   = (const float*)d_in[4];
  const float* bk   = (const float*)d_in[5];
  const float* Wv   = (const float*)d_in[6];
  const float* bv   = (const float*)d_in[7];
  const float* Wo   = (const float*)d_in[8];
  const float* bo   = (const float*)d_in[9];
  float* out = (float*)d_out;

  char*  ws  = (char*)d_ws;
  size_t off = 0;
  auto carve = [&](size_t bytes) -> char* {
    char* p = ws + off;
    off += (bytes + 255) & ~(size_t)255;
    return p;
  };
  bf16_t* Xb  = (bf16_t*)carve((size_t)S_LEN * D_MODEL * sizeof(bf16_t));
  bf16_t* Wqb = (bf16_t*)carve((size_t)D_MODEL * D_MODEL * sizeof(bf16_t));
  bf16_t* Wkb = (bf16_t*)carve((size_t)(N_KV * HEAD_DIM) * D_MODEL * sizeof(bf16_t));
  bf16_t* Wvb = (bf16_t*)carve((size_t)(N_KV * HEAD_DIM) * D_MODEL * sizeof(bf16_t));
  bf16_t* Wob = (bf16_t*)carve((size_t)D_MODEL * D_MODEL * sizeof(bf16_t));
  bf16_t* Qb  = (bf16_t*)carve((size_t)S_LEN * D_MODEL * sizeof(bf16_t));
  bf16_t* Kb  = (bf16_t*)carve((size_t)S_LEN * (N_KV * HEAD_DIM) * sizeof(bf16_t));
  bf16_t* Vtb = (bf16_t*)carve((size_t)(N_KV * HEAD_DIM) * S_LEN * sizeof(bf16_t));
  bf16_t* AOb = (bf16_t*)carve((size_t)S_LEN * D_MODEL * sizeof(bf16_t));

  auto cvt = [&](const float* src, bf16_t* dst, int n) {
    f32_to_bf16_kernel<<<(n + 255) / 256, 256, 0, stream>>>(src, dst, n);
  };
  cvt(x,  Xb,  S_LEN * D_MODEL);
  cvt(Wq, Wqb, D_MODEL * D_MODEL);
  cvt(Wk, Wkb, N_KV * HEAD_DIM * D_MODEL);
  cvt(Wv, Wvb, N_KV * HEAD_DIM * D_MODEL);
  cvt(Wo, Wob, D_MODEL * D_MODEL);

  const int KVD = N_KV * HEAD_DIM;  // 256
  // Q projection: bf16 row-major out
  gemm_bf16_wmma<<<(S_LEN / BM) * (D_MODEL / BN), 128, 0, stream>>>(
      Xb, Wqb, bq, Qb, S_LEN, D_MODEL, D_MODEL, 0, D_MODEL);
  // K projection: bf16 row-major out
  gemm_bf16_wmma<<<(S_LEN / BM) * (KVD / BN), 128, 0, stream>>>(
      Xb, Wkb, bk, Kb, S_LEN, KVD, D_MODEL, 0, KVD);
  // V projection: bf16 TRANSPOSED out -> Vt[256, S]
  gemm_bf16_wmma<<<(S_LEN / BM) * (KVD / BN), 128, 0, stream>>>(
      Xb, Wvb, bv, Vtb, S_LEN, KVD, D_MODEL, 1, S_LEN);

  // attention: one wave per (head, 16-query block)
  swa_attn_wmma<<<N_HEADS * (S_LEN / 16), 32, 0, stream>>>(Qb, Kb, Vtb, sink, AOb);

  // output projection: fp32 out
  gemm_bf16_wmma<<<(S_LEN / BM) * (D_MODEL / BN), 128, 0, stream>>>(
      AOb, Wob, bo, out, S_LEN, D_MODEL, D_MODEL, 2, D_MODEL);
}
